// AnswerModule_65446711656764
// MI455X (gfx1250) — matrix operand order repaired and checked by reading, not compiled, not run on hardware
//
#include <hip/hip_runtime.h>

// CDNA5 / gfx1250: wave32, WMMA 16x16x32 f16 -> f32
typedef __attribute__((ext_vector_type(16))) _Float16 v16h;
typedef __attribute__((ext_vector_type(8)))  _Float16 v8h;
typedef __attribute__((ext_vector_type(4)))  _Float16 v4h;
typedef __attribute__((ext_vector_type(8)))  float    v8f;

#define B_SZ   1024
#define N_OBJ  64
#define DO_    512
#define DA_    512
#define DC_    512
#define DIN2   514     // D_O + 1 indicator + 1 folded bias column (==1.0)
#define VOCAB_ 32000

#define TP     40      // f16 tile row pitch: 32 + 8 pad; 80 B row = 16B-aligned chunks
#define WXP    68      // wx LDS row pitch (f16)

__device__ __forceinline__ void cvt_store4(_Float16* dst, float4 v) {
  v4h h;
  h.x = (_Float16)v.x; h.y = (_Float16)v.y;
  h.z = (_Float16)v.z; h.w = (_Float16)v.w;
  *(v4h*)dst = h;                       // 8B ds_store_b64
}

__device__ __forceinline__ v16h ld_frag(const _Float16* p) {
  v8h lo = *(const v8h*)(p);            // 16B ds_load_b128
  v8h hi = *(const v8h*)(p + 8);
  v16h r;
#pragma unroll
  for (int j = 0; j < 8; ++j) { r[j] = lo[j]; r[8 + j] = hi[j]; }
  return r;
}

// ---------------------------------------------------------------------------
// Kernel 1: wx[b,i] = sum_n w[b,n] * X_aug[b,n,i]; wx[b,513] = 1.0 (bias fold)
// ---------------------------------------------------------------------------
__global__ __launch_bounds__(256)
void pool_kernel(const float* __restrict__ X,
                 const float* __restrict__ Wt,
                 float* __restrict__ wx) {
  __shared__ float ws[N_OBJ];
  const int b   = blockIdx.x;
  const int tid = threadIdx.x;
  if (tid < N_OBJ) ws[tid] = Wt[b * N_OBJ + tid];
  __syncthreads();
  const float* Xb = X + (size_t)b * N_OBJ * DO_;
  for (int i = tid; i < DIN2; i += 256) {
    float acc = 0.f;
    if (i == 513) {
      acc = 1.0f;
    } else if (i == 0) {
#pragma unroll 8
      for (int n = 0; n < N_OBJ; ++n) acc += ws[n];
    } else {
#pragma unroll 8
      for (int n = 0; n < N_OBJ; ++n) acc += ws[n] * Xb[n * DO_ + (i - 1)];
    }
    wx[(size_t)b * DIN2 + i] = acc;
  }
}

// ---------------------------------------------------------------------------
// Kernel 2: ans[b,o] = sum_{i,c} wx[b,i] * W_slice_i[o,c] * q[b,c]
//   (i = 0..512 -> W_op[i]; i = 513 -> W_bias, wx==1)
// GEMM M=1024, N=512, K=514*512; A[b,(i,c)] = wx[b,i]*q[b,c] built in regs.
// Block 256(M)x128(N), 16 waves (4Mx4N), wave 64x32. Split-K over i (8 chunks),
// f32 atomicAdd into ans. W tile f16-staged, double-buffered LDS pipeline.
// ---------------------------------------------------------------------------
#define ICH 65

__global__ __launch_bounds__(512)
void condgemm_kernel(const float* __restrict__ wx,
                     const float* __restrict__ q,
                     const float* __restrict__ W_op,
                     const float* __restrict__ W_bias,
                     float* __restrict__ ans) {
  __shared__ _Float16 tileh[2][128 * TP];   // W slice [o 128][c 32] f16, dbl-buf
  __shared__ _Float16 wxs[256 * WXP];       // wx chunk [b_local 256][i_local]

  const int tid  = threadIdx.x;
  const int wave = tid >> 5;
  const int lane = tid & 31;
  const int half = lane >> 4;
  const int nl   = lane & 15;
  const int wm   = wave & 3;
  const int wn   = wave >> 2;

  const int bB0 = blockIdx.x * 256;
  const int b0  = bB0 + wm * 64;
  const int o0  = blockIdx.y * 128;
  const int i_begin = blockIdx.z * ICH;
  const int i_end   = min(DIN2, i_begin + ICH);
  const int icount  = i_end - i_begin;

  // stage wx chunk (coalesced along i), converted to f16
  {
    const int bl  = tid >> 1;
    const int il0 = (tid & 1) * 33;
    const float* src = wx + (size_t)(bB0 + bl) * DIN2 + i_begin;
    for (int il = il0; il < il0 + 33; ++il)
      if (il < icount) wxs[bl * WXP + il] = (_Float16)src[il];
  }

  v8f acc[4][2];
#pragma unroll
  for (int mf = 0; mf < 4; ++mf)
#pragma unroll
    for (int nf = 0; nf < 2; ++nf)
#pragma unroll
      for (int r = 0; r < 8; ++r) acc[mf][nf][r] = 0.f;

  // this thread's two staging quads within the 128x32 tile
  const int row0 = tid >> 3,           c40 = (tid & 7) * 4;
  const int row1 = (tid + 512) >> 3,   c41 = ((tid + 512) & 7) * 4;

  for (int cs = 0; cs < 16; ++cs) {
    const int c0 = cs * 32;
    // q fragments for this c slice (A layout: lane=M row, half picks K half)
    v16h qh[4];
#pragma unroll
    for (int mf = 0; mf < 4; ++mf) {
      const float* qr = q + (size_t)(b0 + mf * 16 + nl) * DC_ + c0;
#pragma unroll
      for (int j = 0; j < 8; ++j) {
        qh[mf][j]     = (_Float16)qr[half * 16 + j];
        qh[mf][8 + j] = (_Float16)qr[half * 16 + 8 + j];
      }
    }

    // pipeline prologue: prefetch tile(i_begin) into registers
    float4 r0, r1;
    {
      const float* Ws = (i_begin < 513) ? W_op + (size_t)i_begin * DA_ * DC_ : W_bias;
      r0 = *(const float4*)(Ws + (size_t)(o0 + row0) * DC_ + c0 + c40);
      r1 = *(const float4*)(Ws + (size_t)(o0 + row1) * DC_ + c0 + c41);
    }
    int p = 0;
    for (int i = i_begin; i < i_end; ++i) {
      _Float16* tp = tileh[p];
      cvt_store4(tp + row0 * TP + c40, r0);
      cvt_store4(tp + row1 * TP + c41, r1);
      __syncthreads();

      // prefetch next i while computing current
      if (i + 1 < i_end) {
        const int in = i + 1;
        const float* Ws = (in < 513) ? W_op + (size_t)in * DA_ * DC_ : W_bias;
        r0 = *(const float4*)(Ws + (size_t)(o0 + row0) * DC_ + c0 + c40);
        r1 = *(const float4*)(Ws + (size_t)(o0 + row1) * DC_ + c0 + c41);
      }

      v16h bf[2];
#pragma unroll
      for (int nf = 0; nf < 2; ++nf)
        bf[nf] = ld_frag(tp + (wn * 32 + nf * 16 + nl) * TP + half * 16);

      // build ALL A fragments first (independent regs, packed f16 muls),
      // then WMMA burst: no WMMA->VALU WAR nops between mf groups
      v16h av[4];
#pragma unroll
      for (int mf = 0; mf < 4; ++mf) {
        const _Float16 wsH = wxs[(wm * 64 + mf * 16 + nl) * WXP + (i - i_begin)];
        v16h wsv;
#pragma unroll
        for (int j = 0; j < 16; ++j) wsv[j] = wsH;
        av[mf] = qh[mf] * wsv;          // 8x v_pk_mul_f16
      }
#pragma unroll
      for (int mf = 0; mf < 4; ++mf)
#pragma unroll
        for (int nf = 0; nf < 2; ++nf) {
          acc[mf][nf] = __builtin_amdgcn_wmma_f32_16x16x32_f16(
              false, av[mf], false, bf[nf], (short)0, acc[mf][nf], false, false);
        }
      p ^= 1;
    }
    __syncthreads();   // buffers safe for next cs's first store
  }

  // split-K reduction into ans
#pragma unroll
  for (int mf = 0; mf < 4; ++mf)
#pragma unroll
    for (int nf = 0; nf < 2; ++nf)
#pragma unroll
      for (int r = 0; r < 8; ++r) {
        const int row = b0 + mf * 16 + half * 8 + r;
        const int col = o0 + wn * 32 + nf * 16 + nl;
        atomicAdd(&ans[(size_t)row * DA_ + col], acc[mf][nf][r]);
      }
}

// ---------------------------------------------------------------------------
// Kernel 3: logits = ans @ E_a^T  (M=1024, K=512, N=32000)
// Block 128x128, 8 waves (2Mx4N). E and ans tiles f16-staged, double-buffered.
// ---------------------------------------------------------------------------
__global__ __launch_bounds__(256)
void logits_kernel(const float* __restrict__ ans,
                   const float* __restrict__ E,
                   float* __restrict__ out) {
  __shared__ _Float16 et[2][128 * TP];      // E slice [v 128][k 32] f16
  __shared__ _Float16 at[2][128 * TP];      // ans slice [b 128][k 32] f16

  const int tid  = threadIdx.x;
  const int wave = tid >> 5;
  const int lane = tid & 31;
  const int half = lane >> 4;
  const int nl   = lane & 15;
  const int wm   = wave >> 2;               // 0..1
  const int wn   = wave & 3;                // 0..3

  const int bB0 = blockIdx.x * 128;
  const int b0  = bB0 + wm * 64;
  const int v0  = blockIdx.y * 128;

  v8f acc[4][2];
#pragma unroll
  for (int mf = 0; mf < 4; ++mf)
#pragma unroll
    for (int nf = 0; nf < 2; ++nf)
#pragma unroll
      for (int r = 0; r < 8; ++r) acc[mf][nf][r] = 0.f;

  int rowq[4], c4q[4];
#pragma unroll
  for (int t = 0; t < 4; ++t) {
    const int e = tid + t * 256;
    rowq[t] = e >> 3;
    c4q[t]  = (e & 7) * 4;
  }

  // prologue: prefetch k-step 0
  float4 re[4], ra[4];
#pragma unroll
  for (int t = 0; t < 4; ++t) {
    re[t] = *(const float4*)(E   + (size_t)(v0  + rowq[t]) * DA_ + c4q[t]);
    ra[t] = *(const float4*)(ans + (size_t)(bB0 + rowq[t]) * DA_ + c4q[t]);
  }

  int p = 0;
  for (int ks = 0; ks < 16; ++ks) {
    _Float16* ep = et[p];
    _Float16* ap = at[p];
#pragma unroll
    for (int t = 0; t < 4; ++t) {
      cvt_store4(ep + rowq[t] * TP + c4q[t], re[t]);
      cvt_store4(ap + rowq[t] * TP + c4q[t], ra[t]);
    }
    __syncthreads();

    if (ks + 1 < 16) {
      const int k0n = (ks + 1) * 32;
#pragma unroll
      for (int t = 0; t < 4; ++t) {
        re[t] = *(const float4*)(E   + (size_t)(v0  + rowq[t]) * DA_ + k0n + c4q[t]);
        ra[t] = *(const float4*)(ans + (size_t)(bB0 + rowq[t]) * DA_ + k0n + c4q[t]);
      }
    }

    v16h bf[2];
#pragma unroll
    for (int nf = 0; nf < 2; ++nf)
      bf[nf] = ld_frag(ep + (wn * 32 + nf * 16 + nl) * TP + half * 16);

    // load all A fragments, then WMMA burst
    v16h av[4];
#pragma unroll
    for (int mf = 0; mf < 4; ++mf) {
      const _Float16* as = ap + (wm * 64 + mf * 16 + nl) * TP;
      v8h lo = *(const v8h*)(as + half * 16);
      v8h hi = *(const v8h*)(as + half * 16 + 8);
#pragma unroll
      for (int j = 0; j < 8; ++j) { av[mf][j] = lo[j]; av[mf][8 + j] = hi[j]; }
    }
#pragma unroll
    for (int mf = 0; mf < 4; ++mf)
#pragma unroll
      for (int nf = 0; nf < 2; ++nf) {
        acc[mf][nf] = __builtin_amdgcn_wmma_f32_16x16x32_f16(
            false, av[mf], false, bf[nf], (short)0, acc[mf][nf], false, false);
      }
    p ^= 1;
    __syncthreads();
  }

#pragma unroll
  for (int mf = 0; mf < 4; ++mf)
#pragma unroll
    for (int nf = 0; nf < 2; ++nf)
#pragma unroll
      for (int r = 0; r < 8; ++r) {
        const int row = b0 + mf * 16 + half * 8 + r;
        const int col = v0 + wn * 32 + nf * 16 + nl;
        out[(size_t)row * VOCAB_ + col] = acc[mf][nf][r];
      }
}

// ---------------------------------------------------------------------------
extern "C" void kernel_launch(void* const* d_in, const int* in_sizes, int n_in,
                              void* d_out, int out_size, void* d_ws, size_t ws_size,
                              hipStream_t stream) {
  (void)in_sizes; (void)n_in; (void)out_size; (void)ws_size;
  const float* X      = (const float*)d_in[0];
  const float* Xw     = (const float*)d_in[1];
  const float* q      = (const float*)d_in[2];
  const float* W_op   = (const float*)d_in[3];
  const float* W_bias = (const float*)d_in[4];
  const float* E_a    = (const float*)d_in[5];
  float* out = (float*)d_out;

  float* wx  = (float*)d_ws;                    // [1024, 514]
  float* ans = wx + (size_t)B_SZ * DIN2;        // [1024, 512]

  hipMemsetAsync(ans, 0, (size_t)B_SZ * DA_ * sizeof(float), stream);
  pool_kernel<<<dim3(B_SZ), dim3(256), 0, stream>>>(X, Xw, wx);
  condgemm_kernel<<<dim3(4, 4, 8), dim3(512), 0, stream>>>(wx, q, W_op, W_bias, ans);
  logits_kernel<<<dim3(8, 250), dim3(256), 0, stream>>>(ans, E_a, out);
}
